// Layers_LSTM_37632503447567
// MI455X (gfx1250) — matrix operand order repaired and checked by reading, not compiled, not run on hardware
//
#include <hip/hip_runtime.h>
#include <hip/hip_bf16.h>
#include <stdint.h>
#include <stddef.h>

// ---------------------------------------------------------------------------
// LSTM, 2 layers, L=512, B=64, C=H=1024 on MI455X (gfx1250, wave32).
//   Phase A (per layer): x_proj = in @ Wih^T + b
//       LDS double-buffered bf16 WMMA GEMM, async global->LDS copies.
//   Phase B (per layer): persistent scan kernel, one grid barrier per step.
//       h tile staged in LDS via async copy (shared by all 8 waves of block),
//       all 4 gates per (16x16) tile held in registers, cell update in-reg.
// ---------------------------------------------------------------------------

#define L_SEQ 512
#define BATCH 64
#define HID   1024
#define G4    4096   // 4*HID

typedef __attribute__((ext_vector_type(16))) __bf16 v16bf;
typedef __attribute__((ext_vector_type(8)))  __bf16 v8bf;
typedef __attribute__((ext_vector_type(8)))  float  v8f;

#define WMMA_BF16(a, b, c) \
    __builtin_amdgcn_wmma_f32_16x16x32_bf16(false, (a), false, (b), (short)0, (c), false, false)

// ---- CDNA5 async global->LDS copy (ASYNCcnt-tracked) ----------------------
#if __has_builtin(__builtin_amdgcn_global_load_async_to_lds_b128)
#define HAVE_ASYNC_LDS 1
#endif

typedef int v4i_vs __attribute__((vector_size(4 * sizeof(int))));
typedef __attribute__((address_space(1))) v4i_vs* gptr_b128;
typedef __attribute__((address_space(3))) v4i_vs* lptr_b128;

__device__ __forceinline__ void async_cp16(const void* g, void* l) {
#ifdef HAVE_ASYNC_LDS
    __builtin_amdgcn_global_load_async_to_lds_b128((gptr_b128)g, (lptr_b128)l, 0, 0);
#else
    *reinterpret_cast<uint4*>(l) = *reinterpret_cast<const uint4*>(g);  // ds_store_b128 fallback
#endif
}

#if __has_builtin(__builtin_amdgcn_s_wait_asynccnt)
#define WAIT_ASYNC(n) __builtin_amdgcn_s_wait_asynccnt(n)
#else
#define WAIT_ASYNC(n) asm volatile("s_wait_asynccnt %0" ::"i"(n) : "memory")
#endif

// ---------------------------------------------------------------------------
__device__ __forceinline__ uint16_t f2bf(float f) {
    uint32_t u = __float_as_uint(f);
    return (uint16_t)((u + 0x7FFFu + ((u >> 16) & 1u)) >> 16);  // round-to-nearest-even
}

// A fragment (16xK=32 bf16): lane = (row&15) + 16*hi.
// hi=0 lanes hold K {0..7, 16..23}; hi=1 lanes hold K {8..15, 24..31}.
__device__ __forceinline__ v16bf frag_a(const uint16_t* __restrict__ base,
                                        int ldk, int row, int k, int hi) {
    const __bf16* p = reinterpret_cast<const __bf16*>(base + (size_t)row * ldk + k);
    v8bf lo = *reinterpret_cast<const v8bf*>(p + 8 * hi);
    v8bf hh = *reinterpret_cast<const v8bf*>(p + 16 + 8 * hi);
    v16bf a;
#pragma unroll
    for (int i = 0; i < 8; ++i) { a[i] = lo[i]; a[i + 8] = hh[i]; }
    return a;
}

// B fragment (K=32 x 16 bf16) from K-contiguous weight row (B = W^T):
// lane n (n<16) holds K {0..7, 16..23} of column n, lane n+16 holds K {8..15, 24..31}.
__device__ __forceinline__ v16bf frag_b(const uint16_t* __restrict__ base,
                                        int ldk, int wrow, int k, int hi) {
    const __bf16* p = reinterpret_cast<const __bf16*>(base + (size_t)wrow * ldk + k);
    v8bf lo = *reinterpret_cast<const v8bf*>(p + 8 * hi);
    v8bf hh = *reinterpret_cast<const v8bf*>(p + 16 + 8 * hi);
    v16bf b;
#pragma unroll
    for (int i = 0; i < 8; ++i) { b[i] = lo[i]; b[i + 8] = hh[i]; }
    return b;
}

// ---------------------------------------------------------------------------
__global__ void f32_to_bf16_kernel(const float* __restrict__ src,
                                   uint16_t* __restrict__ dst, int n) {
    int i = blockIdx.x * blockDim.x + threadIdx.x;
    if (i < n) dst[i] = f2bf(src[i]);
}

// ---------------------------------------------------------------------------
// x_proj GEMM: out[M=32768, N=4096] = A[M,1024] @ W[4096,1024]^T + bias.
// Block: 256 thr = 8 waves as 4(M) x 2(N); block tile 64M x 128N; K-panel 64.
// Double-buffered LDS (padded rows), async copies, windowed asynccnt waits.
// ---------------------------------------------------------------------------
#define KP   64     // K panel
#define PAD  72     // LDS row stride in halfs (144B: conflict-free 16-row reads)

__global__ __launch_bounds__(256) void xproj_gemm_kernel(
    const uint16_t* __restrict__ A,     // [32768, 1024] bf16
    const uint16_t* __restrict__ W,     // [4096, 1024]  bf16
    const float*    __restrict__ bias,  // [4096]
    float*          __restrict__ out)   // [32768, 4096]
{
    __shared__ uint16_t sA[2][64 * PAD];    // 18.0 KB
    __shared__ uint16_t sB[2][128 * PAD];   // 36.9 KB

    const int tid  = threadIdx.x;
    const int lane = tid & 31;
    const int wv   = tid >> 5;
    const int hi   = lane >> 4;
    const int l15  = lane & 15;
    const int mblk = blockIdx.y * 64;
    const int nblk = blockIdx.x * 128;
    const int arow = (wv >> 1) * 16 + l15;  // local A row for this lane
    const int nw   = wv & 1;

    // Stage one K-panel: A 64x64 (512 chunks), B 128x64 (1024 chunks); 6 ops/thread.
    auto issue = [&](int buf, int k0) {
#pragma unroll
        for (int j = 0; j < 2; ++j) {
            int c = tid + 256 * j, row = c >> 3, col = (c & 7) * 8;
            async_cp16(A + (size_t)(mblk + row) * HID + k0 + col, &sA[buf][row * PAD + col]);
        }
#pragma unroll
        for (int j = 0; j < 4; ++j) {
            int c = tid + 256 * j, row = c >> 3, col = (c & 7) * 8;
            async_cp16(W + (size_t)(nblk + row) * HID + k0 + col, &sB[buf][row * PAD + col]);
        }
    };

    issue(0, 0);
    v8f acc[4] = {};
    for (int kt = 0; kt < HID / KP; ++kt) {
        const int buf = kt & 1;
        if (kt + 1 < HID / KP) { issue(buf ^ 1, (kt + 1) * KP); WAIT_ASYNC(6); }
        else                   { WAIT_ASYNC(0); }
        __syncthreads();
#pragma unroll
        for (int kk = 0; kk < KP; kk += 32) {
            v16bf a = frag_a(&sA[buf][0], PAD, arow, kk, hi);
#pragma unroll
            for (int s = 0; s < 4; ++s) {
                v16bf b = frag_b(&sB[buf][0], PAD, nw * 64 + s * 16 + l15, kk, hi);
                acc[s] = WMMA_BF16(a, b, acc[s]);
            }
        }
        __syncthreads();  // all waves done with buf before it is refilled
    }

    const int m0 = mblk + (wv >> 1) * 16;
    const int n0 = nblk + nw * 64;
#pragma unroll
    for (int s = 0; s < 4; ++s) {
        const int col = n0 + s * 16 + l15;
        const float bv = bias[col];
#pragma unroll
        for (int r = 0; r < 8; ++r) {
            const int m = m0 + r + 8 * hi;       // C/D layout: M = r + 8*hi, N = lane&15
            out[(size_t)m * G4 + col] = acc[s][r] + bv;
        }
    }
}

// ---------------------------------------------------------------------------
// Persistent recurrent scan. 32 blocks x 256 threads = 256 waves.
// All 8 waves of a block share one batch tile (16 rows of h): staged once per
// step into LDS by async copy. Wave task: 16 batch rows x 16 hidden cols, four
// gate accumulators, in-register cell update, one device barrier per step.
// ---------------------------------------------------------------------------
#define HPAD 1032   // LDS row stride in halfs for the h tile

__global__ __launch_bounds__(256) void lstm_scan_kernel(
    const float*    __restrict__ xproj,    // [512*64, 4096]
    const uint16_t* __restrict__ Whh,      // [4096, 1024] bf16
    uint16_t*       __restrict__ hbuf,     // [2][64*1024] bf16 double buffer
    float*          __restrict__ cbuf,     // [64*1024]
    uint16_t*       __restrict__ out_bf16, // layer0: next-layer input; else null
    float*          __restrict__ out_f32,  // layer1: final outs f32; else null
    float*          __restrict__ out_h,    // [64*1024] final h (f32)
    float*          __restrict__ out_c,    // [64*1024] final c (f32)
    unsigned*       __restrict__ barrier_cnt,
    int nblocks)
{
    __shared__ uint16_t sH[16 * HPAD];     // 33 KB: this block's batch tile of h

    const int tid  = threadIdx.x;
    const int lane = tid & 31;
    const int wv   = tid >> 5;
    const int hi   = lane >> 4;
    const int l15  = lane & 15;
    const int task = blockIdx.x * 8 + wv;    // 0..255
    const int m0   = (task >> 6) * 16;       // batch tile (same for all waves in block)
    const int hcol0= (task & 63) * 16;       // hidden column tile
    const int col  = hcol0 + l15;

    for (int t = 0; t < L_SEQ; ++t) {
        const uint16_t* hin  = hbuf + (size_t)(t & 1) * (BATCH * HID) + (size_t)m0 * HID;
        uint16_t*       hout = hbuf + (size_t)((t + 1) & 1) * (BATCH * HID);

        // ---- stage h tile (16 x 1024 bf16 = 32KB) into LDS: 8 chunks/thread ----
#pragma unroll
        for (int j = 0; j < 8; ++j) {
            int c = tid + 256 * j, row = c >> 7, kcol = (c & 127) * 8;
            async_cp16(hin + (size_t)row * HID + kcol, &sH[row * HPAD + kcol]);
        }
        WAIT_ASYNC(0);
        __syncthreads();

        // ---- gates = h @ Whh^T for this wave's tile, all 4 gates ----
        v8f acc[4] = {};                     // i, f, g, o
        for (int k = 0; k < HID; k += 32) {
            v16bf a = frag_a(sH, HPAD, l15, k, hi);
#pragma unroll
            for (int g = 0; g < 4; ++g) {
                v16bf b = frag_b(Whh, HID, g * HID + hcol0 + l15, k, hi);
                acc[g] = WMMA_BF16(a, b, acc[g]);
            }
        }

        // ---- in-register LSTM cell update ----
        const size_t xbase = (size_t)t * BATCH * G4;
#pragma unroll
        for (int r = 0; r < 8; ++r) {
            const int m = m0 + r + 8 * hi;
            const float* xp = xproj + xbase + (size_t)m * G4;
            float gi = acc[0][r] + xp[col];
            float gf = acc[1][r] + xp[HID + col];
            float gg = acc[2][r] + xp[2 * HID + col];
            float go = acc[3][r] + xp[3 * HID + col];
            gi = 1.f / (1.f + __expf(-gi));
            gf = 1.f / (1.f + __expf(-gf));
            gg = tanhf(gg);
            go = 1.f / (1.f + __expf(-go));
            const int ci = m * HID + col;
            const float cnew = gf * cbuf[ci] + gi * gg;
            const float hnew = go * tanhf(cnew);
            cbuf[ci] = cnew;
            const uint16_t hb = f2bf(hnew);
            hout[ci] = hb;
            if (out_bf16) out_bf16[(size_t)(t * BATCH + m) * HID + col] = hb;
            if (out_f32)  out_f32 [(size_t)(t * BATCH + m) * HID + col] = hnew;
            if (t == L_SEQ - 1) { out_h[ci] = hnew; out_c[ci] = cnew; }
        }

        if (t != L_SEQ - 1) {
            // ---- device-scope barrier (monotonic counter, release/acquire) ----
            __threadfence();
            __syncthreads();
            if (tid == 0) {
                __hip_atomic_fetch_add(barrier_cnt, 1u, __ATOMIC_RELEASE,
                                       __HIP_MEMORY_SCOPE_AGENT);
                const unsigned target = (unsigned)(t + 1) * (unsigned)nblocks;
                while (__hip_atomic_load(barrier_cnt, __ATOMIC_ACQUIRE,
                                         __HIP_MEMORY_SCOPE_AGENT) < target) {
                    __builtin_amdgcn_s_sleep(1);
                }
            }
            __syncthreads();   // also guards LDS h-tile reuse next step
        }
    }
}

// ---------------------------------------------------------------------------
// Workspace layout (bytes)
// ---------------------------------------------------------------------------
static const size_t OFF_CNT   = 0;
static const size_t OFF_HBUF  = 256;
static const size_t OFF_CBUF  = OFF_HBUF + 2u * BATCH * HID * 2;
static const size_t OFF_WIH   = OFF_CBUF + (size_t)BATCH * HID * 4;
static const size_t OFF_WHH   = OFF_WIH + 2ull * G4 * HID * 2;
static const size_t OFF_IN    = OFF_WHH + 2ull * G4 * HID * 2;
static const size_t OFF_XPROJ = OFF_IN + (size_t)L_SEQ * BATCH * HID * 2;

extern "C" void kernel_launch(void* const* d_in, const int* in_sizes, int n_in,
                              void* d_out, int out_size, void* d_ws, size_t ws_size,
                              hipStream_t stream) {
    const float* x   = (const float*)d_in[0];  // [512,64,1024]
    const float* sh  = (const float*)d_in[1];  // [2,64,1024]
    const float* sc  = (const float*)d_in[2];  // [2,64,1024]
    const float* wih = (const float*)d_in[3];  // [2,4096,1024]
    const float* whh = (const float*)d_in[4];  // [2,4096,1024]
    const float* bb  = (const float*)d_in[5];  // [2,4096]

    char* ws = (char*)d_ws;
    unsigned* cnt   = (unsigned*)(ws + OFF_CNT);
    uint16_t* hbuf  = (uint16_t*)(ws + OFF_HBUF);
    float*    cbuf  = (float*)   (ws + OFF_CBUF);
    uint16_t* wihb  = (uint16_t*)(ws + OFF_WIH);
    uint16_t* whhb  = (uint16_t*)(ws + OFF_WHH);
    uint16_t* inb   = (uint16_t*)(ws + OFF_IN);
    float*    xproj = (float*)   (ws + OFF_XPROJ);

    float* outs = (float*)d_out;                       // [512,64,1024]
    float* outH = outs + (size_t)L_SEQ * BATCH * HID;  // [2,64,1024]
    float* outC = outH + 2u * BATCH * HID;             // [2,64,1024]

    // bf16 conversions
    {
        int n = L_SEQ * BATCH * HID;
        f32_to_bf16_kernel<<<(n + 255) / 256, 256, 0, stream>>>(x, inb, n);
        n = 2 * G4 * HID;
        f32_to_bf16_kernel<<<(n + 255) / 256, 256, 0, stream>>>(wih, wihb, n);
        f32_to_bf16_kernel<<<(n + 255) / 256, 256, 0, stream>>>(whh, whhb, n);
    }

    for (int l = 0; l < 2; ++l) {
        const int n = BATCH * HID;
        f32_to_bf16_kernel<<<(n + 255) / 256, 256, 0, stream>>>(sh + (size_t)l * n, hbuf, n);
        (void)hipMemcpyAsync(cbuf, sc + (size_t)l * n, (size_t)n * sizeof(float),
                             hipMemcpyDeviceToDevice, stream);

        // Phase A: x_proj for all timesteps (parallel, LDS-pipelined GEMM)
        xproj_gemm_kernel<<<dim3(G4 / 128, (L_SEQ * BATCH) / 64), 256, 0, stream>>>(
            inb, wihb + (size_t)l * G4 * HID, bb + (size_t)l * G4, xproj);

        // Phase B: persistent recurrent scan
        (void)hipMemsetAsync(cnt, 0, sizeof(unsigned), stream);
        lstm_scan_kernel<<<32, 256, 0, stream>>>(
            xproj, whhb + (size_t)l * G4 * HID, hbuf, cbuf,
            (l == 0) ? inb  : (uint16_t*)nullptr,
            (l == 1) ? outs : (float*)nullptr,
            outH + (size_t)l * n, outC + (size_t)l * n,
            cnt, 32);
    }
}
